// GAT_67525475828069
// MI455X (gfx1250) — matrix operand order repaired and checked by reading, not compiled, not run on hardware
//
#include <hip/hip_runtime.h>
#include <hip/hip_bf16.h>
#include <cstdint>

typedef __attribute__((ext_vector_type(16))) __bf16   v16bf;
typedef __attribute__((ext_vector_type(8)))  __bf16   v8bf;
typedef __attribute__((ext_vector_type(8)))  float    v8f;
typedef __attribute__((ext_vector_type(8)))  unsigned v8u;

constexpr int   Bb   = 8;
constexpr int   Nn   = 1024;
constexpr int   FIN  = 512;
constexpr int   FOUT = 256;
constexpr int   Hh   = 8;
constexpr float ALPHA = 0.2f;
constexpr float NEGV  = -1e9f;

__device__ __forceinline__ unsigned pack_bf16(float a, float b) {
  unsigned short lo = __builtin_bit_cast(unsigned short, (__bf16)a);
  unsigned short hi = __builtin_bit_cast(unsigned short, (__bf16)b);
  return (unsigned)lo | ((unsigned)hi << 16);
}

// ---------------------------------------------------------------------------
// Kernel 0a: convert x -> bf16, pre-swizzled into WMMA A-fragment order.
// Each 32-K chunk kept in natural order [K0..7 | K8..15 | K16..23 | K24..31] so that a
// lane's A fragment (v16bf) is one contiguous 32-byte load at offset 16*g.
// One thread per 32-element chunk.
// ---------------------------------------------------------------------------
__global__ void __launch_bounds__(256)
cvt_x_kernel(const float* __restrict__ x, __bf16* __restrict__ xsw)
{
  const size_t idx = (size_t)blockIdx.x * 256 + threadIdx.x;   // chunk id
  const float4* p = (const float4*)(x + idx * 32);
  float4 f0 = p[0], f1 = p[1], f2 = p[2], f3 = p[3];
  float4 f4 = p[4], f5 = p[5], f6 = p[6], f7 = p[7];
  uint4* q = (uint4*)(xsw + idx * 32);
  uint4 s0 = { pack_bf16(f0.x,f0.y), pack_bf16(f0.z,f0.w),
               pack_bf16(f1.x,f1.y), pack_bf16(f1.z,f1.w) };   // K0..7
  uint4 s1 = { pack_bf16(f4.x,f4.y), pack_bf16(f4.z,f4.w),
               pack_bf16(f5.x,f5.y), pack_bf16(f5.z,f5.w) };   // K16..23
  uint4 s2 = { pack_bf16(f2.x,f2.y), pack_bf16(f2.z,f2.w),
               pack_bf16(f3.x,f3.y), pack_bf16(f3.z,f3.w) };   // K8..15
  uint4 s3 = { pack_bf16(f6.x,f6.y), pack_bf16(f6.z,f6.w),
               pack_bf16(f7.x,f7.y), pack_bf16(f7.z,f7.w) };   // K24..31
  q[0] = s0; q[1] = s2; q[2] = s1; q[3] = s3;
}

// ---------------------------------------------------------------------------
// Kernel 0b: convert W -> bf16 (natural layout; B fragments are contiguous).
// One thread per 8 elements.
// ---------------------------------------------------------------------------
__global__ void __launch_bounds__(256)
cvt_w_kernel(const float* __restrict__ W, __bf16* __restrict__ wbf)
{
  const size_t idx = (size_t)blockIdx.x * 256 + threadIdx.x;
  const float4* p = (const float4*)(W + idx * 8);
  float4 a = p[0], b = p[1];
  uint4 s = { pack_bf16(a.x,a.y), pack_bf16(a.z,a.w),
              pack_bf16(b.x,b.y), pack_bf16(b.z,b.w) };
  *(uint4*)(wbf + idx * 8) = s;
}

// ---------------------------------------------------------------------------
// Kernel 1: feature transform  h = x @ W^T  (bf16 WMMA, f32 accum).
// Each wave: 16(rows) x 64(cols) tile -> 4 C frags; A loaded once per K-chunk.
// Outputs: Ht bf16 [h][b][o][m] (m fastest), f1/f2 via shfl-reduce + atomics.
// ---------------------------------------------------------------------------
__global__ void __launch_bounds__(256)
gat_feat_kernel(const __bf16* __restrict__ xsw, const __bf16* __restrict__ wbf,
                const float* __restrict__ a1, const float* __restrict__ a2,
                __bf16* __restrict__ Ht, float* __restrict__ f1,
                float* __restrict__ f2)
{
  const int lane = threadIdx.x & 31;
  const int wid  = blockIdx.x * 8 + (threadIdx.x >> 5);  // 0 .. 16383
  const int ot4 = wid & 3;          // group of 4 F_OUT tiles (64 cols)
  const int nt  = (wid >> 2) & 63;  // N tile
  const int b   = (wid >> 8) & 7;
  const int h   = (wid >> 11) & 7;

  const int g   = lane >> 4;
  const int r   = lane & 15;
  const int row = nt * 16 + r;
  const int ob  = ot4 * 64;

  const __bf16* __restrict__ xrow = xsw + (size_t)(b * Nn + row) * FIN;
  const __bf16* __restrict__ wr0 =
      wbf + (size_t)(h * FOUT + ob + r) * FIN;           // t=0 row; t stride = 16*FIN

  v8f c[4] = {{}, {}, {}, {}};

#pragma unroll 2
  for (int kb = 0; kb < FIN; kb += 32) {
    // A fragment: one contiguous 32B load (pre-swizzled chunk, half g)
    v16bf af = *(const v16bf*)(xrow + kb + 16 * g);
#pragma unroll
    for (int t = 0; t < 4; ++t) {
      // B fragment: col = ob + t*16 + r ; K = kb + 16g + {0..15} contiguous
      v16bf bf = *(const v16bf*)(wr0 + (size_t)t * 16 * FIN + kb + 16 * g);
      c[t] = __builtin_amdgcn_wmma_f32_16x16x32_bf16(false, af, false, bf,
                                                     (short)0, c[t], false, false);
    }
  }

  // Store transposed bf16 Ht and accumulate f1/f2 partials.
  float s1[8], s2[8];
#pragma unroll
  for (int e = 0; e < 8; ++e) { s1[e] = 0.0f; s2[e] = 0.0f; }

#pragma unroll
  for (int t = 0; t < 4; ++t) {
    const int o = ob + t * 16 + r;
    uint4 st;
    st.x = pack_bf16(c[t][0], c[t][1]);
    st.y = pack_bf16(c[t][2], c[t][3]);
    st.z = pack_bf16(c[t][4], c[t][5]);
    st.w = pack_bf16(c[t][6], c[t][7]);
    size_t hofs = ((size_t)((h * Bb + b) * FOUT + o)) * Nn + nt * 16 + 8 * g;
    *(uint4*)(Ht + hofs) = st;

    const float a1v = a1[h * FOUT + o];
    const float a2v = a2[h * FOUT + o];
#pragma unroll
    for (int e = 0; e < 8; ++e) {
      s1[e] += c[t][e] * a1v;
      s2[e] += c[t][e] * a2v;
    }
  }

#pragma unroll
  for (int m = 1; m < 16; m <<= 1) {
#pragma unroll
    for (int e = 0; e < 8; ++e) {
      s1[e] += __shfl_xor(s1[e], m);
      s2[e] += __shfl_xor(s2[e], m);
    }
  }
  if (r == 0) {
    const int base = (h * Bb + b) * Nn + nt * 16 + 8 * g;
#pragma unroll
    for (int e = 0; e < 8; ++e) {
      atomicAdd(&f1[base + e], s1[e]);
      atomicAdd(&f2[base + e], s2[e]);
    }
  }
}

// ---------------------------------------------------------------------------
// Kernel 2: flash-style attention + aggregation.
// Block = (h, b, 16-row tile); 4 waves, wave w covers output cols [64w,64w+64).
// Per 32-wide m chunk: e = mask(leaky(f1+f2)), online softmax, p(bf16)
// A-fragment x Ht B-fragments via WMMA. Final: /L, ELU, atomic-add over heads.
// ---------------------------------------------------------------------------
__global__ void __launch_bounds__(128)
gat_attn_kernel(const int* __restrict__ adj, const __bf16* __restrict__ Ht,
                const float* __restrict__ f1, const float* __restrict__ f2,
                float* __restrict__ acc)
{
  const int lane = threadIdx.x & 31;
  const int w    = threadIdx.x >> 5;       // 0..3
  const int rt = blockIdx.x & 63;
  const int b  = (blockIdx.x >> 6) & 7;
  const int h  = (blockIdx.x >> 9) & 7;

  const int g   = lane >> 4;
  const int r   = lane & 15;
  const int row = rt * 16 + r;             // this lane's attention row

  const float f1row = f1[(h * Bb + b) * Nn + row];
  const float* __restrict__ f2p  = f2 + (h * Bb + b) * Nn;
  const int*   __restrict__ adjp = adj + (size_t)(b * Nn + row) * Nn;
  const __bf16* __restrict__ htb = Ht + (size_t)((h * Bb + b) * FOUT) * Nn;
  const int obase = w * 64;

  v8f   c[4] = {{}, {}, {}, {}};
  float M = -INFINITY, L = 0.0f;

  for (int mb = 0; mb < Nn; mb += 32) {
    const int k0 = mb + 16 * g;  // lane's first m index in this chunk
    float4 fa = *(const float4*)(f2p + k0);
    float4 fb = *(const float4*)(f2p + k0 + 4);
    float4 fc = *(const float4*)(f2p + k0 + 8);
    float4 fd = *(const float4*)(f2p + k0 + 12);
    int4 qa = *(const int4*)(adjp + k0);
    int4 qb = *(const int4*)(adjp + k0 + 4);
    int4 qc = *(const int4*)(adjp + k0 + 8);
    int4 qd = *(const int4*)(adjp + k0 + 12);

    auto mk = [&](float f2x, int a) -> float {
      float e = f1row + f2x;
      e = fmaxf(e, ALPHA * e);            // leaky relu (ALPHA < 1)
      return (a == 0) ? NEGV : e;         // adjacency mask
    };
    float ev[16];
    ev[0]  = mk(fa.x, qa.x); ev[1]  = mk(fa.y, qa.y);
    ev[2]  = mk(fa.z, qa.z); ev[3]  = mk(fa.w, qa.w);
    ev[4]  = mk(fb.x, qb.x); ev[5]  = mk(fb.y, qb.y);
    ev[6]  = mk(fb.z, qb.z); ev[7]  = mk(fb.w, qb.w);
    ev[8]  = mk(fc.x, qc.x); ev[9]  = mk(fc.y, qc.y);
    ev[10] = mk(fc.z, qc.z); ev[11] = mk(fc.w, qc.w);
    ev[12] = mk(fd.x, qd.x); ev[13] = mk(fd.y, qd.y);
    ev[14] = mk(fd.z, qd.z); ev[15] = mk(fd.w, qd.w);

    // Online softmax: lanes r and r+16 jointly own row r's 32 values.
    float cm = ev[0];
#pragma unroll
    for (int i = 1; i < 16; ++i) cm = fmaxf(cm, ev[i]);
    cm = fmaxf(cm, __shfl_xor(cm, 16));
    const float Mn = fmaxf(M, cm);
    const float sc = __expf(M - Mn);

    float rs = 0.0f;
    v8u pu;
#pragma unroll
    for (int i = 0; i < 8; ++i) {
      float p0 = __expf(ev[2 * i]     - Mn);
      float p1 = __expf(ev[2 * i + 1] - Mn);
      rs += p0 + p1;
      pu[i] = pack_bf16(p0, p1);
    }
    v16bf af = __builtin_bit_cast(v16bf, pu);
    rs += __shfl_xor(rs, 16);
    L = L * sc + rs;
    M = Mn;

    // Rescale accumulators: scale for row (8g+e) lives in lane (8g+e).
#pragma unroll
    for (int e = 0; e < 8; ++e) {
      float se = __shfl(sc, e + 8 * g);
#pragma unroll
      for (int t = 0; t < 4; ++t) c[t][e] *= se;
    }

    // 4 WMMAs: B fragment = Ht tile (32 m x 16 o), contiguous in m.
#pragma unroll
    for (int t = 0; t < 4; ++t) {
      const __bf16* hp = htb + (size_t)(obase + t * 16 + r) * Nn + mb + 16 * g;
      v16bf bf = *(const v16bf*)hp;
      c[t] = __builtin_amdgcn_wmma_f32_16x16x32_bf16(false, af, false, bf,
                                                     (short)0, c[t], false, false);
    }
  }

  // Finalize: divide by L (per-row), ELU, accumulate across heads.
#pragma unroll
  for (int e = 0; e < 8; ++e) {
    float Le  = __shfl(L, e + 8 * g);
    float inv = 1.0f / Le;
    const int orow = rt * 16 + 8 * g + e;
#pragma unroll
    for (int t = 0; t < 4; ++t) {
      float v = c[t][e] * inv;
      v = (v > 0.0f) ? v : (__expf(v) - 1.0f);   // ELU
      atomicAdd(&acc[(size_t)(b * Nn + orow) * FOUT + obase + t * 16 + r], v);
    }
  }
}

// ---------------------------------------------------------------------------
// Kernel 3: log_softmax over F_OUT = 256.  One wave per (b, n) row.
// ---------------------------------------------------------------------------
__global__ void __launch_bounds__(256)
gat_lsm_kernel(const float* __restrict__ acc, float* __restrict__ out)
{
  const int lane  = threadIdx.x & 31;
  const int rowid = blockIdx.x * 8 + (threadIdx.x >> 5);  // 0 .. B*N-1
  const float* p = acc + (size_t)rowid * FOUT;

  float v[8];
  float m = -INFINITY;
#pragma unroll
  for (int j = 0; j < 8; ++j) {
    v[j] = p[lane + 32 * j];
    m = fmaxf(m, v[j]);
  }
#pragma unroll
  for (int o = 16; o; o >>= 1) m = fmaxf(m, __shfl_xor(m, o));

  float s = 0.0f;
#pragma unroll
  for (int j = 0; j < 8; ++j) s += __expf(v[j] - m);
#pragma unroll
  for (int o = 16; o; o >>= 1) s += __shfl_xor(s, o);

  const float ls = m + __logf(s);
  float* q = out + (size_t)rowid * FOUT;
#pragma unroll
  for (int j = 0; j < 8; ++j) q[lane + 32 * j] = v[j] - ls;
}

// ---------------------------------------------------------------------------
extern "C" void kernel_launch(void* const* d_in, const int* in_sizes, int n_in,
                              void* d_out, int out_size, void* d_ws, size_t ws_size,
                              hipStream_t stream) {
  const float* x   = (const float*)d_in[0];
  const int*   adj = (const int*)d_in[1];
  const float* W   = (const float*)d_in[2];
  const float* a1  = (const float*)d_in[3];
  const float* a2  = (const float*)d_in[4];
  float* out = (float*)d_out;

  // Workspace: [ xsw 8MB | wbf 2MB | Ht 32MB | f1 256KB | f2 256KB | acc 8MB ]
  char* ws = (char*)d_ws;
  __bf16* xsw = (__bf16*)ws;
  const size_t xBytes = (size_t)Bb * Nn * FIN * sizeof(__bf16);
  __bf16* wbf = (__bf16*)(ws + xBytes);
  const size_t wBytes = (size_t)Hh * FOUT * FIN * sizeof(__bf16);
  __bf16* Ht = (__bf16*)(ws + xBytes + wBytes);
  const size_t htBytes = (size_t)Hh * Bb * FOUT * Nn * sizeof(__bf16);
  float* f1  = (float*)(ws + xBytes + wBytes + htBytes);
  float* f2  = f1 + (size_t)Hh * Bb * Nn;
  float* acc = f2 + (size_t)Hh * Bb * Nn;
  const size_t zeroBytes =
      (2 * (size_t)Hh * Bb * Nn + (size_t)Bb * Nn * FOUT) * sizeof(float);
  hipMemsetAsync(f1, 0, zeroBytes, stream);

  // 0a: B*N*FIN/32 = 131072 chunks -> 512 blocks
  cvt_x_kernel<<<512, 256, 0, stream>>>(x, xsw);
  // 0b: H*FOUT*FIN/8 = 131072 threads -> 512 blocks
  cvt_w_kernel<<<512, 256, 0, stream>>>(W, wbf);
  // 1: 16384 wave-tiles -> 2048 blocks of 8 waves
  gat_feat_kernel<<<2048, 256, 0, stream>>>(xsw, wbf, a1, a2, Ht, f1, f2);
  // 2: H*B*(N/16) = 4096 blocks of 4 waves
  gat_attn_kernel<<<4096, 128, 0, stream>>>(adj, Ht, f1, f2, acc);
  // 3: B*N rows, one wave each -> 1024 blocks
  gat_lsm_kernel<<<1024, 256, 0, stream>>>(acc, out);
}
